// RNN_53145925320808
// MI455X (gfx1250) — matrix operand order repaired and checked by reading, not compiled, not run on hardware
//
#include <hip/hip_runtime.h>

// ---------------------------------------------------------------------------
// 2-layer ReLU RNN + FC for MI455X (gfx1250), bf16 WMMA + f32 accumulate.
//   B=256, T=512, I=512, H=1024, O=256
// GEMM kernel structure (per block = 8 waves):
//   - block owns one 16-row A slab; staged once into LDS (32KB, padded rows)
//       * bf16 A: GLOBAL_LOAD_ASYNC_TO_LDS_B128 (ASYNCcnt) + s_wait_asynccnt
//       * f32  A: global_load_b128 -> v_cvt_pk_bf16_f32 -> ds_store_b128
//   - each wave computes a 16 x (16*NJ) tile; A fragments via ds_load_b128,
//     B fragments streamed from L2-resident row-major W with unconditional
//     speculative global_prefetch (WGP scope)
//   - v_wmma_f32_16x16x32_bf16, K consumed 32/iter
// ---------------------------------------------------------------------------

typedef __attribute__((ext_vector_type(16))) __bf16 v16bf;
typedef __attribute__((ext_vector_type(8)))  __bf16 v8bf;
typedef __attribute__((ext_vector_type(8)))  float  v8f;

#define BB 256
#define TT 512
#define II 512
#define HH 1024
#define OO 256

__device__ __forceinline__ v8f wmma_bf16(v16bf a, v16bf b, v8f c) {
  // (neg_a, A, neg_b, B, c_mod, C, reuse_a, reuse_b)
  return __builtin_amdgcn_wmma_f32_16x16x32_bf16(false, a, false, b,
                                                 (short)0, c, false, false);
}

__global__ void cvt_f32_bf16(const float* __restrict__ s, __bf16* __restrict__ d, int n) {
  int i = blockIdx.x * blockDim.x + threadIdx.x;
  if (i < n) d[i] = (__bf16)s[i];
}

__global__ void relu_f32_to_bf16(const float* __restrict__ s, __bf16* __restrict__ d, int n) {
  int i = blockIdx.x * blockDim.x + threadIdx.x;
  if (i < n) { float v = s[i]; d[i] = (__bf16)(v > 0.f ? v : 0.f); }
}

// C[M,N] = A[M,K] @ W[N,K]^T (+bias0[N] +bias1[N]) (+pre[M,N]) (relu) -> f32/bf16
// Fragment layouts (A/B: contiguous k order chosen here, same for both; C/D as in ISA):
//   A (16x32): lane L -> row L%16; elem e -> K=(L/16)*16+e
//     => two 16B ds_load_b128 per lane from the LDS slab
//   B (32x16): lane L -> col L%16; elem e -> K=(L/16)*16+e
//     => one contiguous 32B global load per lane from row-major W[N][K]
//   C/D: VGPR r, lane L -> (M = r + 8*(L/16), N = L%16)
template<int NJ, bool A_F32, bool XMAP, bool ADD_PRE, bool RELU, bool OUT_BF16>
__global__ __launch_bounds__(256)
void gemm_wmma(const void* __restrict__ Ap, const __bf16* __restrict__ W,
               const float* __restrict__ pre, const float* __restrict__ bias0,
               const float* __restrict__ bias1, void* __restrict__ Cp,
               int M, int N, int K)
{
  __shared__ __bf16 sA[16 * (HH + 8)];       // 16 rows, +16B pad per row (banking)
  const int SK   = K + 8;                    // padded row stride (elements)
  const int lane = threadIdx.x & 31;
  const int wave = threadIdx.x >> 5;
  const int m0   = blockIdx.y << 4;                // one m-tile per block
  const int n0   = (blockIdx.x * 8 + wave) * 16 * NJ;  // one n-tile per wave
  const int row  = lane & 15;
  const int grp  = lane >> 4;

  // ---- cooperative stage of the 16 x K A slab into LDS ----
  {
    const int chunksPerRow = K >> 3;         // 8 bf16 (16B) per chunk
    const int nChunks = chunksPerRow << 4;
    for (int i = threadIdx.x; i < nChunks; i += 256) {
      const int r  = i / chunksPerRow;
      const int kc = (i - r * chunksPerRow) << 3;
      const int m  = m0 + r;
      // output row m = t*B+b ; x stored [B,T,I] -> row at (b*T + t)*K
      const long ao = XMAP ? ((long)(m & (BB - 1)) * TT + (m >> 8)) * K
                           : (long)m * K;
      if (A_F32) {
        v8f f = *(const v8f*)((const float*)Ap + ao + kc);
        v8bf h;
#pragma unroll
        for (int e = 0; e < 8; ++e) h[e] = (__bf16)f[e];
        *(v8bf*)(sA + r * SK + kc) = h;
      } else {
        const unsigned lds = (unsigned)(uintptr_t)(sA + r * SK + kc);
        const unsigned long long ga =
            (unsigned long long)(uintptr_t)((const __bf16*)Ap + ao + kc);
        asm volatile("global_load_async_to_lds_b128 %0, %1, off"
                     :: "v"(lds), "v"(ga) : "memory");
      }
    }
  }
  if (!A_F32) {
#if __has_builtin(__builtin_amdgcn_s_wait_asynccnt)
    __builtin_amdgcn_s_wait_asynccnt(0);
#else
    asm volatile("s_wait_asynccnt 0x0" ::: "memory");
#endif
  }
  __syncthreads();

  // ---- main K loop ----
  v8f acc[NJ] = {};
  const __bf16* wp[NJ];
#pragma unroll
  for (int j = 0; j < NJ; ++j)
    wp[j] = W + (long)(n0 + j * 16 + row) * K + grp * 16;
  const __bf16* aL = sA + row * SK + grp * 16;

  for (int k0 = 0; k0 < K; k0 += 32) {
    union { v16bf v; v8bf h[2]; } af;
    af.h[0] = *(const v8bf*)(aL + k0);        // ds_load_b128
    af.h[1] = *(const v8bf*)(aL + k0 + 8);    // ds_load_b128
#pragma unroll
    for (int j = 0; j < NJ; ++j) {
      // Speculative prefetch (TH drops bad translations); 128B ahead on the
      // streamed W row, WGP scope so it lands in L0 as well as GL2.
      __builtin_prefetch(wp[j] + k0 + 64, 0, 3);
      v16bf bv = *(const v16bf*)(wp[j] + k0); // 32B global load (L2-resident W)
      acc[j] = wmma_bf16(af.v, bv, acc[j]);
    }
  }

  // ---- epilogue ----
#pragma unroll
  for (int j = 0; j < NJ; ++j) {
    const int n = n0 + j * 16 + row;
    float badd = 0.0f;
    if (bias0) badd += bias0[n];
    if (bias1) badd += bias1[n];
#pragma unroll
    for (int r = 0; r < 8; ++r) {
      const int mm = m0 + r + 8 * grp;
      float v = acc[j][r] + badd;
      if (ADD_PRE) v += pre[(long)mm * N + n];
      if (RELU)    v = v > 0.f ? v : 0.f;
      if (OUT_BF16) ((__bf16*)Cp)[(long)mm * N + n] = (__bf16)v;
      else          ((float*)Cp)[(long)mm * N + n] = v;
    }
  }
}

extern "C" void kernel_launch(void* const* d_in, const int* in_sizes, int n_in,
                              void* d_out, int out_size, void* d_ws, size_t ws_size,
                              hipStream_t stream) {
  const float* x    = (const float*)d_in[0];
  const float* Wih0 = (const float*)d_in[1];
  const float* Whh0 = (const float*)d_in[2];
  const float* bih0 = (const float*)d_in[3];
  const float* bhh0 = (const float*)d_in[4];
  const float* Wih1 = (const float*)d_in[5];
  const float* Whh1 = (const float*)d_in[6];
  const float* bih1 = (const float*)d_in[7];
  const float* bhh1 = (const float*)d_in[8];
  const float* fcw  = (const float*)d_in[9];
  const float* fcb  = (const float*)d_in[10];

  char* wsp = (char*)d_ws;
  size_t off = 0;
  auto carve = [&](size_t bytes) -> void* {
    void* p = wsp + off;
    off += (bytes + 255) & ~(size_t)255;
    return p;
  };
  float*  pre  = (float*) carve((size_t)TT * BB * HH * 4);  // 512 MB, reused per layer
  __bf16* hbuf = (__bf16*)carve((size_t)TT * BB * HH * 2);  // 256 MB, reused per layer
  __bf16* wih0 = (__bf16*)carve((size_t)HH * II * 2);
  __bf16* whh0 = (__bf16*)carve((size_t)HH * HH * 2);
  __bf16* wih1 = (__bf16*)carve((size_t)HH * HH * 2);
  __bf16* whh1 = (__bf16*)carve((size_t)HH * HH * 2);
  __bf16* fcwb = (__bf16*)carve((size_t)OO * HH * 2);

  auto cvt = [&](const float* s, __bf16* d, int n) {
    cvt_f32_bf16<<<(n + 255) / 256, 256, 0, stream>>>(s, d, n);
  };
  cvt(Wih0, wih0, HH * II);
  cvt(Whh0, whh0, HH * HH);
  cvt(Wih1, wih1, HH * HH);
  cvt(Whh1, whh1, HH * HH);
  cvt(fcw,  fcwb, OO * HH);

  const int MBT = BB * TT;     // 131072 rows for the pre-projections
  const int BH  = BB * HH;     // one timestep of hidden state

  const dim3 gPre (HH / 64 / 8, MBT / 16);  // NJ=4: (2, 8192)
  const dim3 gStep(HH / 32 / 8, BB  / 16);  // NJ=2: (4, 16)
  const dim3 gFc  (OO / 16 / 8, BB  / 16);  // NJ=1: (2, 16)

  // --- layer 0 pre-projection: f32 x converted to bf16 on the fly ---
  gemm_wmma<4, true, true, false, false, false><<<gPre, 256, 0, stream>>>(
      x, wih0, nullptr, bih0, bhh0, pre, MBT, HH, II);

  // --- layer 0 recurrent scan ---
  relu_f32_to_bf16<<<BH / 256, 256, 0, stream>>>(pre, hbuf, BH);  // t=0, h_{-1}=0
  for (int t = 1; t < TT; ++t) {
    gemm_wmma<2, false, false, true, true, true><<<gStep, 256, 0, stream>>>(
        hbuf + (size_t)(t - 1) * BH, whh0, pre + (size_t)t * BH,
        nullptr, nullptr, hbuf + (size_t)t * BH, BB, HH, HH);
  }
  // --- layer 1 pre-projection (A = layer-0 hidden, bf16, contiguous [T*B, H]) ---
  gemm_wmma<4, false, false, false, false, false><<<gPre, 256, 0, stream>>>(
      hbuf, wih1, nullptr, bih1, bhh1, pre, MBT, HH, HH);

  // --- layer 1 recurrent scan (overwrites hbuf; pre1 already consumed hbuf) ---
  relu_f32_to_bf16<<<BH / 256, 256, 0, stream>>>(pre, hbuf, BH);
  for (int t = 1; t < TT; ++t) {
    gemm_wmma<2, false, false, true, true, true><<<gStep, 256, 0, stream>>>(
        hbuf + (size_t)(t - 1) * BH, whh1, pre + (size_t)t * BH,
        nullptr, nullptr, hbuf + (size_t)t * BH, BB, HH, HH);
  }
  // --- final FC on last hidden state: [256,1024] @ [256,1024]^T + fc_b ---
  gemm_wmma<1, false, false, false, false, false><<<gFc, 256, 0, stream>>>(
      hbuf + (size_t)(TT - 1) * BH, fcwb, nullptr, fcb, nullptr,
      d_out, BB, OO, HH);
}
